// _AVWGCN_89696097010341
// MI455X (gfx1250) — matrix operand order repaired and checked by reading, not compiled, not run on hardware
//
#include <hip/hip_runtime.h>

// ---------------------------------------------------------------------------
// AVWGCN for MI455X (gfx1250, wave32).
// Heavy path: Y1 = A@X, Y2 = A@Y1 via v_wmma_f32_16x16x32_bf16.
// T2 = 2A^2 - I folded algebraically: out = X*(W0-W2) + Y1*W1 + Y2*(2*W2) + b.
// ---------------------------------------------------------------------------

typedef __attribute__((ext_vector_type(16))) __bf16 v16bf;
typedef __attribute__((ext_vector_type(8)))  __bf16 v8bf;
typedef __attribute__((ext_vector_type(8)))  float  v8f;
typedef __attribute__((ext_vector_type(4)))  float  v4f;

static constexpr int BB = 64;    // batch
static constexpr int NN = 2048;  // nodes
static constexpr int CC = 32;    // channels in = out
static constexpr int DD = 16;    // embed dim

// ---------------------------------------------------------------------------
// Kernel 1: A[n,m] = softmax_m(relu(E[n,:]·E[m,:])), stored row-major bf16.
// One block (256 thr) per row n; row staged in LDS for max/sum passes.
// ---------------------------------------------------------------------------
__global__ __launch_bounds__(256) void k_adj_softmax(
    const float* __restrict__ E, __bf16* __restrict__ Abf) {
  __shared__ float row[NN];
  __shared__ float red[256];
  const int n = blockIdx.x;
  const int tid = threadIdx.x;

  float En[DD];
#pragma unroll
  for (int d = 0; d < DD; ++d) En[d] = E[n * DD + d];  // uniform -> scalar loads

  float lmax = 0.0f;  // relu floor is 0
  for (int m = tid; m < NN; m += 256) {
    const float* Em = E + m * DD;
    float acc = 0.0f;
#pragma unroll
    for (int d = 0; d < DD; ++d) acc += En[d] * Em[d];
    acc = fmaxf(acc, 0.0f);
    row[m] = acc;
    lmax = fmaxf(lmax, acc);
  }
  red[tid] = lmax;
  __syncthreads();
  for (int s = 128; s > 0; s >>= 1) {
    if (tid < s) red[tid] = fmaxf(red[tid], red[tid + s]);
    __syncthreads();
  }
  const float rmax = red[0];
  __syncthreads();

  float lsum = 0.0f;
  for (int m = tid; m < NN; m += 256) {
    float e = __expf(row[m] - rmax);
    row[m] = e;
    lsum += e;
  }
  red[tid] = lsum;
  __syncthreads();
  for (int s = 128; s > 0; s >>= 1) {
    if (tid < s) red[tid] += red[tid + s];
    __syncthreads();
  }
  const float rinv = 1.0f / red[0];
  for (int m = tid; m < NN; m += 256)
    Abf[(size_t)n * NN + m] = (__bf16)(row[m] * rinv);
}

// ---------------------------------------------------------------------------
// Kernel 2: XT[b][c][m] = (bf16) x[b][m][c].  LDS-tiled transpose+convert.
// Block handles 64 m x 32 c for one b.
// ---------------------------------------------------------------------------
__global__ __launch_bounds__(256) void k_transpose_x(
    const float* __restrict__ x, __bf16* __restrict__ XT) {
  __shared__ float tile[CC][64 + 1];
  const int m0 = blockIdx.x * 64;
  const int b = blockIdx.y;
  const int tid = threadIdx.x;

#pragma unroll
  for (int r = 0; r < 8; ++r) {
    int idx = r * 256 + tid;
    int ml = idx >> 5, c = idx & 31;
    tile[c][ml] = x[((size_t)b * NN + m0 + ml) * CC + c];
  }
  __syncthreads();
#pragma unroll
  for (int r = 0; r < 8; ++r) {
    int idx = r * 256 + tid;
    int c = idx >> 6, ml = idx & 63;
    XT[((size_t)b * CC + c) * NN + m0 + ml] = (__bf16)tile[c][ml];
  }
}

// ---------------------------------------------------------------------------
// Kernel 3: effective per-node weights (Chebyshev folded) + bias.
//   W_k[n,i,o] = sum_d E[n,d] * Wpool[d,k,i,o]
//   U[n,0]=W0-W2, U[n,1]=W1, U[n,2]=2*W2;  biasN[n]=E[n]@bias_pool
// ---------------------------------------------------------------------------
__global__ __launch_bounds__(256) void k_weights(
    const float* __restrict__ E, const float* __restrict__ Wpool,
    const float* __restrict__ bias_pool, float* __restrict__ U,
    float* __restrict__ biasN) {
  const int n = blockIdx.x;
  const int tid = threadIdx.x;
  float En[DD];
#pragma unroll
  for (int d = 0; d < DD; ++d) En[d] = E[n * DD + d];

  for (int f = tid; f < CC * CC; f += 256) {  // f = i*32 + o
    float w0 = 0.f, w1 = 0.f, w2 = 0.f;
#pragma unroll
    for (int d = 0; d < DD; ++d) {
      const float e = En[d];
      const float* wp = Wpool + (size_t)d * 3 * CC * CC + f;
      w0 += e * wp[0];
      w1 += e * wp[CC * CC];
      w2 += e * wp[2 * CC * CC];
    }
    float* un = U + (size_t)n * 3 * CC * CC + f;
    un[0]           = w0 - w2;
    un[CC * CC]     = w1;
    un[2 * CC * CC] = 2.0f * w2;
  }
  if (tid < CC) {
    float bs = 0.f;
#pragma unroll
    for (int d = 0; d < DD; ++d) bs += En[d] * bias_pool[d * CC + tid];
    biasN[n * CC + tid] = bs;
  }
}

// ---------------------------------------------------------------------------
// Kernel 4: Y^T = (A @ X_b)^T via v_wmma_f32_16x16x32_bf16.
//   A   : (NN x NN) bf16 row-major  (WMMA A-operand layout: 2x16B row chunks)
//   BT  : (BB x CC x NN) bf16       (RHS transposed = WMMA B-operand layout)
//   OT  : (BB x CC x NN)            bf16 (OUT_BF16=1) or f32 (OUT_BF16=0)
// One wave computes a 32(M) x 32(Ncol) block: 2x2 tiles, 4 wmma / K-step.
// D-tile rows land contiguous per lane -> direct b128 stores of transposed out.
// ---------------------------------------------------------------------------
__device__ __forceinline__ v16bf load_a_op(const __bf16* p) {
  // elements 0..7 at p, 8..15 at p+16  (K = 8h..8h+7 and 16+8h..16+8h+7)
  v8bf lo = *(const v8bf*)p;
  v8bf hi = *(const v8bf*)(p + 16);
  v16bf r;
#pragma unroll
  for (int e = 0; e < 8; ++e) { r[e] = lo[e]; r[e + 8] = hi[e]; }
  return r;
}
__device__ __forceinline__ v16bf load_b_op(const __bf16* p) {
  // elements 0..7 at p, 8..15 at p+16  (K = 8h..8h+7 and 16+8h..16+8h+7)
  v8bf lo = *(const v8bf*)p;
  v8bf hi = *(const v8bf*)(p + 16);
  v16bf r;
#pragma unroll
  for (int e = 0; e < 8; ++e) { r[e] = lo[e]; r[e + 8] = hi[e]; }
  return r;
}
__device__ __forceinline__ v8f wmma_bf16(v16bf a, v16bf b, v8f c) {
  return __builtin_amdgcn_wmma_f32_16x16x32_bf16(
      false, a, false, b, (short)0, c, false, false);
}

template <int OUT_BF16>
__global__ __launch_bounds__(128) void k_gemm(
    const __bf16* __restrict__ A, const __bf16* __restrict__ BT,
    void* __restrict__ OT) {
  const int lane = threadIdx.x & 31;
  const int wave = threadIdx.x >> 5;
  const int mt = (blockIdx.x * 4 + wave) * 32;  // 32 output rows (nodes)
  const int b = blockIdx.y;
  const int h = lane >> 4;
  const int l = lane & 15;

  const __bf16* Ar0 = A + (size_t)(mt + l) * NN;
  const __bf16* Ar1 = A + (size_t)(mt + 16 + l) * NN;
  const __bf16* Br0 = BT + ((size_t)b * CC + l) * NN;
  const __bf16* Br1 = BT + ((size_t)b * CC + 16 + l) * NN;

  v8f acc00 = {}, acc01 = {}, acc10 = {}, acc11 = {};

  for (int k0 = 0; k0 < NN; k0 += 32) {
    __builtin_prefetch(Ar0 + k0 + 256, 0, 1);
    __builtin_prefetch(Br0 + k0 + 256, 0, 1);
    v16bf a0 = load_a_op(Ar0 + k0 + 8 * h);
    v16bf a1 = load_a_op(Ar1 + k0 + 8 * h);
    v16bf b0 = load_b_op(Br0 + k0 + 8 * h);
    v16bf b1 = load_b_op(Br1 + k0 + 8 * h);
    acc00 = wmma_bf16(a0, b0, acc00);
    acc01 = wmma_bf16(a0, b1, acc01);
    acc10 = wmma_bf16(a1, b0, acc10);
    acc11 = wmma_bf16(a1, b1, acc11);
  }

  // D layout: lane holds rows m = 8h + r (r=0..7) of one column -> contiguous.
  v8f accs[4] = {acc00, acc01, acc10, acc11};
#pragma unroll
  for (int t = 0; t < 4; ++t) {
    const int mth = t >> 1;          // which 16-row half of the 32-row block
    const int ct = t & 1;            // which 16-column half
    const size_t col = (size_t)b * CC + ct * 16 + l;
    const size_t moff = (size_t)mt + mth * 16 + 8 * h;
    if (OUT_BF16) {
      v8bf o;
#pragma unroll
      for (int r = 0; r < 8; ++r) o[r] = (__bf16)accs[t][r];
      *(v8bf*)((__bf16*)OT + col * NN + moff) = o;
    } else {
      v4f lo, hi;
#pragma unroll
      for (int r = 0; r < 4; ++r) { lo[r] = accs[t][r]; hi[r] = accs[t][r + 4]; }
      *(v4f*)((float*)OT + col * NN + moff) = lo;
      *(v4f*)((float*)OT + col * NN + moff + 4) = hi;
    }
  }
}

// ---------------------------------------------------------------------------
// Kernel 5: out[b,n,o] = bias[n,o] + sum_i X[b,n,i]*U0[n,i,o]
//                        + Y1T[b,i,n]*U1[n,i,o] + Y2T[b,i,n]*U2[n,i,o]
// Block per (n, 8 batches); U[n] (12 KB) cached in LDS.
// ---------------------------------------------------------------------------
__global__ __launch_bounds__(256) void k_final(
    const float* __restrict__ x, const __bf16* __restrict__ Y1T,
    const float* __restrict__ Y2T, const float* __restrict__ U,
    const float* __restrict__ biasN, float* __restrict__ out) {
  __shared__ float sU[3 * CC * CC];
  __shared__ float sb[CC];
  const int n = blockIdx.x;
  const int tid = threadIdx.y * 32 + threadIdx.x;

  for (int f = tid; f < 3 * CC * CC; f += 256) sU[f] = U[(size_t)n * 3 * CC * CC + f];
  if (tid < CC) sb[tid] = biasN[n * CC + tid];
  __syncthreads();

  const int o = threadIdx.x;
  const int b = blockIdx.y * 8 + threadIdx.y;

  const float*  xr = x + ((size_t)b * NN + n) * CC;
  const __bf16* y1 = Y1T + (size_t)b * CC * NN + n;
  const float*  y2 = Y2T + (size_t)b * CC * NN + n;

  float acc = sb[o];
#pragma unroll
  for (int i = 0; i < CC; ++i) {
    acc += xr[i] * sU[i * CC + o];
    acc += (float)y1[(size_t)i * NN] * sU[CC * CC + i * CC + o];
    acc += y2[(size_t)i * NN] * sU[2 * CC * CC + i * CC + o];
  }
  out[((size_t)b * NN + n) * CC + o] = acc;
}

// ---------------------------------------------------------------------------
// Host orchestration
// ---------------------------------------------------------------------------
extern "C" void kernel_launch(void* const* d_in, const int* in_sizes, int n_in,
                              void* d_out, int out_size, void* d_ws, size_t ws_size,
                              hipStream_t stream) {
  const float* x     = (const float*)d_in[0];  // (64, 2048, 32)
  const float* E     = (const float*)d_in[1];  // (2048, 16)
  const float* Wpool = (const float*)d_in[2];  // (16, 3, 32, 32)
  const float* bpool = (const float*)d_in[3];  // (16, 32)
  float* out = (float*)d_out;                  // (64, 2048, 32)

  char* ws = (char*)d_ws;
  const size_t SZ_ABF = (size_t)NN * NN * 2;        //  8 MB
  const size_t SZ_XT  = (size_t)BB * CC * NN * 2;   //  8 MB
  const size_t SZ_Y2  = (size_t)BB * CC * NN * 4;   // 16 MB
  const size_t SZ_U   = (size_t)NN * 3 * CC * CC * 4; // 24 MB
  __bf16* Abf   = (__bf16*)(ws);
  __bf16* XT    = (__bf16*)(ws + SZ_ABF);
  __bf16* Y1T   = (__bf16*)(ws + SZ_ABF + SZ_XT);
  float*  Y2T   = (float*)(ws + SZ_ABF + 2 * SZ_XT);
  float*  U     = (float*)(ws + SZ_ABF + 2 * SZ_XT + SZ_Y2);
  float*  biasN = (float*)(ws + SZ_ABF + 2 * SZ_XT + SZ_Y2 + SZ_U);

  k_adj_softmax<<<NN, 256, 0, stream>>>(E, Abf);
  k_transpose_x<<<dim3(NN / 64, BB), 256, 0, stream>>>(x, XT);
  k_weights<<<NN, 256, 0, stream>>>(E, Wpool, bpool, U, biasN);
  k_gemm<1><<<dim3(NN / 128, BB), 128, 0, stream>>>(Abf, XT, (void*)Y1T);   // Y1 = A@X
  k_gemm<0><<<dim3(NN / 128, BB), 128, 0, stream>>>(Abf, Y1T, (void*)Y2T);  // Y2 = A@Y1
  k_final<<<dim3(NN, BB / 8), dim3(32, 8), 0, stream>>>(x, Y1T, Y2T, U, biasN, out);
}
